// DHHT_47107201303163
// MI455X (gfx1250) — matrix operand order repaired and checked by reading, not compiled, not run on hardware
//
#include <hip/hip_runtime.h>
#include <hip/hip_bf16.h>

// ---------------------------------------------------------------------------
// MI455X (gfx1250): all large GEMMs via v_wmma_f32_16x16x32_bf16 (fp32 accum).
// LDS tiles are staged in fragment-major layout so each WMMA operand is one
// 32-byte vector load per lane (2x ds_load_b128). Tile fills exploit that a
// 16-element K-segment of a source row is contiguous in the fragment image,
// so K-major sources are copied with full-width vector loads/stores.
// ---------------------------------------------------------------------------

typedef __attribute__((ext_vector_type(16))) __bf16 v16bf;
typedef __attribute__((ext_vector_type(8)))  __bf16 v8bf;
typedef __attribute__((ext_vector_type(8)))  float  v8f;
typedef __attribute__((ext_vector_type(4)))  float  v4f;

#define NN   100000L
#define DD   128
#define EE   256
#define SCL  0.08838834764831845f   // 128^-0.5
#define EPSA 1e-8f

// Fragment-major chunk: 32 lanes x 16 halfs = 512 halfs = 1024 B per 16x32 tile.
#define FRAG 512

// A element (m, k), k in [0,32): lane 0-15: M=lane, K in {0..15};
// lane 16-31: M=lane-16, K in {16..31}.
__device__ __forceinline__ int a_slot(int m, int k) {
  int lane = m + ((k & 16) ? 16 : 0);
  int h    = k & 15;
  return lane * 16 + h;
}

__device__ __forceinline__ v16bf load_frag(const __bf16* chunk, int lane) {
  return *(const v16bf*)(chunk + lane * 16);     // 32 B -> 2x ds_load_b128
}

__device__ __forceinline__ v8f wmma_bf16(v16bf a, v16bf b, v8f c) {
  return __builtin_amdgcn_wmma_f32_16x16x32_bf16(false, a, false, b, (short)0, c,
                                                 false, false);
}

__device__ __forceinline__ float wave_max(float v) {
#pragma unroll
  for (int o = 16; o > 0; o >>= 1) v = fmaxf(v, __shfl_xor(v, o, 32));
  return v;
}
__device__ __forceinline__ float wave_sum(float v) {
#pragma unroll
  for (int o = 16; o > 0; o >>= 1) v += __shfl_xor(v, o, 32);
  return v;
}

// Fill one B-fragment 16-K-segment from 16 fp32 (K-contiguous source row).
// dst32 = &smB[fragidx*FRAG + lane*16]; one 32B LDS store.
__device__ __forceinline__ void b_fill16_f32(__bf16* dst32, const float* src) {
  const v8f* s = (const v8f*)src;
  v8f lo = s[0], hi = s[1];
  v16bf val;
#pragma unroll
  for (int j = 0; j < 8; ++j) { val[j] = (__bf16)lo[j]; val[8 + j] = (__bf16)hi[j]; }
  *(v16bf*)dst32 = val;
}

// ---------------------------------------------------------------------------
// K1: edges = LN(mu + exp(ls)*noise); q = relu(edges @ Wq^T + bq)   [E x 128]
__global__ void k_edge_prep(const float* __restrict__ mu, const float* __restrict__ ls,
                            const float* __restrict__ noise,
                            const float* __restrict__ g, const float* __restrict__ b,
                            const float* __restrict__ Wq, const float* __restrict__ bq,
                            float* __restrict__ edges_out, __bf16* __restrict__ q_out) {
  __shared__ float row[128];
  __shared__ float red[128];
  int t = threadIdx.x;
  long e = blockIdx.x;
  float v = mu[e * 128 + t] + __expf(ls[e * 128 + t]) * noise[e * 128 + t];
  red[t] = v; __syncthreads();
  for (int o = 64; o > 0; o >>= 1) { if (t < o) red[t] += red[t + o]; __syncthreads(); }
  float mean = red[0] * (1.f / 128.f); __syncthreads();
  float d = v - mean;
  red[t] = d * d; __syncthreads();
  for (int o = 64; o > 0; o >>= 1) { if (t < o) red[t] += red[t + o]; __syncthreads(); }
  float rstd = rsqrtf(red[0] * (1.f / 128.f) + 1e-5f);
  float xn = d * rstd * g[t] + b[t];
  row[t] = xn;
  edges_out[e * 128 + t] = xn;
  __syncthreads();
  float acc = bq[t];
#pragma unroll 4
  for (int k = 0; k < 128; ++k) acc += row[k] * Wq[t * 128 + k];
  q_out[e * 128 + t] = (__bf16)fmaxf(acc, 0.f);
}

// ---------------------------------------------------------------------------
// K2: x_n = LN(x); kf = relu(x_n Wk^T), vf = relu(x_n Wv^T), q2 = x_n Wq^T
__global__ void k_node_gemm(const float* __restrict__ x,
                            const float* __restrict__ Wk, const float* __restrict__ bk,
                            const float* __restrict__ Wv, const float* __restrict__ bv,
                            const float* __restrict__ Wq, const float* __restrict__ bq,
                            const float* __restrict__ g, const float* __restrict__ bb,
                            __bf16* __restrict__ kf, __bf16* __restrict__ vf,
                            __bf16* __restrict__ q2, __bf16* __restrict__ xb) {
  __shared__ __align__(32) __bf16 smX[4 * FRAG];        // 16x128 A, 4 K-chunks
  __shared__ __align__(32) __bf16 smW[32 * FRAG];       // 128x128 B, 8 nt x 4 kc
  __shared__ float part[256];
  __shared__ float stat[32];
  int t = threadIdx.x, lane = t & 31, w = t >> 5;
  int r = t >> 4, cb = t & 15;
  long row = (long)blockIdx.x * 16 + r;

  float xv[8]; float s = 0.f;
#pragma unroll
  for (int j = 0; j < 8; ++j) { xv[j] = x[row * 128 + cb + 16 * j]; s += xv[j]; }
  part[t] = s; __syncthreads();
  if (cb == 0) { float m = 0; for (int i = 0; i < 16; ++i) m += part[r * 16 + i]; stat[r] = m * (1.f / 128.f); }
  __syncthreads();
  float mean = stat[r]; float sq = 0.f;
#pragma unroll
  for (int j = 0; j < 8; ++j) { float d = xv[j] - mean; sq += d * d; }
  part[t] = sq; __syncthreads();
  if (cb == 0) { float m = 0; for (int i = 0; i < 16; ++i) m += part[r * 16 + i]; stat[16 + r] = rsqrtf(m * (1.f / 128.f) + 1e-5f); }
  __syncthreads();
  float rstd = stat[16 + r];
#pragma unroll
  for (int j = 0; j < 8; ++j) {
    int c = cb + 16 * j;
    smX[(c >> 5) * FRAG + a_slot(r, c & 31)] =
        (__bf16)((xv[j] - mean) * rstd * g[c] + bb[c]);
    xb[row * 128 + c] = (__bf16)xv[j];
  }
  __syncthreads();

  v16bf afr[4];
#pragma unroll
  for (int kc = 0; kc < 4; ++kc) afr[kc] = load_frag(smX + kc * FRAG, lane);

  const float* Ws[3] = {Wk, Wv, Wq};
  const float* bs[3] = {bk, bv, bq};
  __bf16* outs[3]    = {kf, vf, q2};
  for (int gi = 0; gi < 3; ++gi) {
    const float* W = Ws[gi];
    // B fill: 128 rows(n) x 8 K-groups of 16; each group = 64B gl load + 32B ds store
#pragma unroll
    for (int i = t; i < 1024; i += 256) {
      int n = i >> 3, gidx = i & 7;
      int ln = (n & 15) + ((gidx & 1) ? 16 : 0);
      b_fill16_f32(smW + ((n >> 4) * 4 + (gidx >> 1)) * FRAG + ln * 16,
                   W + n * 128 + gidx * 16);
    }
    __syncthreads();
    v8f acc = {};
#pragma unroll
    for (int kc = 0; kc < 4; ++kc)
      acc = wmma_bf16(afr[kc], load_frag(smW + (w * 4 + kc) * FRAG, lane), acc);
    int n = w * 16 + (lane & 15);
    int mb = (lane & 16) ? 8 : 0;
    float bias = bs[gi][n];
#pragma unroll
    for (int rr = 0; rr < 8; ++rr) {
      float vv = acc[rr] + bias;
      if (gi != 2) vv = fmaxf(vv, 0.f);
      outs[gi][((long)blockIdx.x * 16 + rr + mb) * 128 + n] = (__bf16)vv;
    }
    __syncthreads();
  }
}

// ---------------------------------------------------------------------------
// K3: dots = q @ kf^T * SCALE   [E x N], grid(ceil(N/128), E/16), block(256)
__global__ void k_dots(const __bf16* __restrict__ qe, const __bf16* __restrict__ kf,
                       float* __restrict__ dots) {
  __shared__ __align__(32) __bf16 smQ[4 * FRAG];        // 16x128 A
  __shared__ __align__(32) __bf16 smK[32 * FRAG];       // 128x128 B
  int t = threadIdx.x, lane = t & 31, w = t >> 5;
  int m0 = blockIdx.y * 16;
  long n_base = (long)blockIdx.x * 128;
  {  // A fill: 16 rows x 16 groups of 8 -> one 16B gl load + 16B ds store each
    int m = t >> 4, gi = t & 15;
    v8bf val = *(const v8bf*)(qe + ((long)m0 + m) * 128 + gi * 8);
    int ln = m + ((gi & 2) ? 16 : 0);
    int h  = (gi & 1) * 8;
    *(v8bf*)(smQ + (gi >> 2) * FRAG + ln * 16 + h) = val;
  }
  // B fill: kf rows are K-contiguous -> 32B gl load + 32B ds store per 16 elems
#pragma unroll
  for (int i = t; i < 1024; i += 256) {
    int n = i >> 3, gi = i & 7;
    long rw = n_base + n; if (rw >= NN) rw = NN - 1;
    v16bf val = *(const v16bf*)(kf + rw * 128 + gi * 16);
    int ln = (n & 15) + ((gi & 1) ? 16 : 0);
    *(v16bf*)(smK + ((n >> 4) * 4 + (gi >> 1)) * FRAG + ln * 16) = val;
  }
  __syncthreads();
  v16bf afr[4];
#pragma unroll
  for (int kc = 0; kc < 4; ++kc) afr[kc] = load_frag(smQ + kc * FRAG, lane);
  v8f acc = {};
#pragma unroll
  for (int kc = 0; kc < 4; ++kc)
    acc = wmma_bf16(afr[kc], load_frag(smK + (w * 4 + kc) * FRAG, lane), acc);
  int nn = w * 16 + (lane & 15);
  long col = n_base + nn;
  int mb = (lane & 16) ? 8 : 0;
  if (col < NN)
#pragma unroll
    for (int rr = 0; rr < 8; ++rr)
      dots[(long)(m0 + rr + mb) * NN + col] = acc[rr] * SCL;
}

// ---------------------------------------------------------------------------
// K4: per edge-row stats: max, 1/sumexp, top-32 threshold (bisection on dots;
//     softmax is monotone), 1/(kept mass + EPS*cnt + 1e-9).  grid(E), block(256)
__global__ void k_row_stats(const float* __restrict__ dots, float* __restrict__ stats) {
  __shared__ float red[256];
  __shared__ float red2[256];
  __shared__ float bc[3];
  int t = threadIdx.x;
  const float* rowp = dots + (long)blockIdx.x * NN;

  float mx = -3.4e38f, mn = 3.4e38f;
  for (long i = t; i < NN; i += 256) { float d = rowp[i]; mx = fmaxf(mx, d); mn = fminf(mn, d); }
  red[t] = mx; red2[t] = mn; __syncthreads();
  for (int o = 128; o > 0; o >>= 1) {
    if (t < o) { red[t] = fmaxf(red[t], red[t + o]); red2[t] = fminf(red2[t], red2[t + o]); }
    __syncthreads();
  }
  if (t == 0) { bc[0] = red[0]; bc[1] = red2[0]; }
  __syncthreads();
  float rmax = bc[0], rmin = bc[1];

  float se = 0.f;
  for (long i = t; i < NN; i += 256) se += __expf(rowp[i] - rmax);
  red[t] = se; __syncthreads();
  for (int o = 128; o > 0; o >>= 1) { if (t < o) red[t] += red[t + o]; __syncthreads(); }
  if (t == 0) bc[2] = red[0];
  __syncthreads();
  float invden = 1.f / bc[2];

  float lo = rmin, hi = rmax + 1.f;
  for (int it = 0; it < 32; ++it) {
    float mid = 0.5f * (lo + hi);
    int cnt = 0;
    for (long i = t; i < NN; i += 256) cnt += (rowp[i] >= mid);
    red[t] = (float)cnt; __syncthreads();
    for (int o = 128; o > 0; o >>= 1) { if (t < o) red[t] += red[t + o]; __syncthreads(); }
    float total = red[0];
    __syncthreads();
    if (total >= 32.f) lo = mid; else hi = mid;
  }
  float thr = lo;

  float ks = 0.f; float kc = 0.f;
  for (long i = t; i < NN; i += 256) {
    float d = rowp[i];
    if (d >= thr) { ks += __expf(d - rmax) * invden; kc += 1.f; }
  }
  red[t] = ks; red2[t] = kc; __syncthreads();
  for (int o = 128; o > 0; o >>= 1) {
    if (t < o) { red[t] += red[t + o]; red2[t] += red2[t + o]; }
    __syncthreads();
  }
  if (t == 0) {
    long e = blockIdx.x;
    stats[e * 4 + 0] = rmax;
    stats[e * 4 + 1] = invden;
    stats[e * 4 + 2] = thr;
    stats[e * 4 + 3] = 1.f / (red[0] + EPSA * red2[0] + 1e-9f);
  }
}

// ---------------------------------------------------------------------------
// K5: updates = masked_attn @ vf  [E x 128]; grid(E/16), block(256)
__global__ void k_updates(const float* __restrict__ dots, const float* __restrict__ stats,
                          const __bf16* __restrict__ vf, float* __restrict__ upd) {
  __shared__ __align__(32) __bf16 smA[FRAG];            // 16x32 attn tile
  __shared__ __align__(32) __bf16 smB[8 * FRAG];        // 32x128 vf chunk
  __shared__ float st[64];
  int t = threadIdx.x, lane = t & 31, w = t >> 5;
  int e0 = blockIdx.x * 16;
  if (t < 64) st[t] = stats[e0 * 4 + t];
  __syncthreads();
  v8f acc = {};
  for (long nc = 0; nc < NN; nc += 32) {
    for (int i = t; i < 512; i += 256) {
      int r = i >> 5, k = i & 31;
      float d  = dots[(long)(e0 + r) * NN + nc + k];
      float sm = __expf(d - st[r * 4 + 0]) * st[r * 4 + 1];
      float a  = (d >= st[r * 4 + 2]) ? (sm + EPSA) * st[r * 4 + 3] : 0.f;
      smA[a_slot(r, k)] = (__bf16)a;
    }
    // vf chunk: vector global read (16B), scatter 8 ds_store_b16
#pragma unroll
    for (int i = t; i < 512; i += 256) {
      int k = i >> 4, cg = i & 15;
      v8bf val = *(const v8bf*)(vf + (nc + k) * 128 + cg * 8);
      int nt = cg >> 1;
      int h = k & 15;
      int lane0 = (cg & 1) * 8 + ((k & 16) ? 16 : 0);
      __bf16* dst = smB + nt * FRAG + h;
#pragma unroll
      for (int j = 0; j < 8; ++j) dst[(lane0 + j) * 16] = val[j];
    }
    if (nc + 32 < NN) {
      __builtin_prefetch(&dots[(long)e0 * NN + nc + 32], 0, 1);   // global_prefetch_b8
      __builtin_prefetch(&vf[(nc + 32) * 128], 0, 1);
    }
    __syncthreads();
    acc = wmma_bf16(load_frag(smA, lane), load_frag(smB + w * FRAG, lane), acc);
    __syncthreads();
  }
  int n = w * 16 + (lane & 15), mb = (lane & 16) ? 8 : 0;
#pragma unroll
  for (int rr = 0; rr < 8; ++rr)
    upd[(long)(e0 + rr + mb) * 128 + n] = acc[rr];
}

// ---------------------------------------------------------------------------
// K6: edge MLP + k2 + ew (tiny); also zeroes De accumulator. grid(E), block(128)
__global__ void k_edge_mlp(const float* __restrict__ edges, const float* __restrict__ upd,
                           const float* __restrict__ w1, const float* __restrict__ b1,
                           const float* __restrict__ w2, const float* __restrict__ b2,
                           const float* __restrict__ Wk, const float* __restrict__ bk,
                           const float* __restrict__ conv_w, const float* __restrict__ conv_b,
                           __bf16* __restrict__ k2, __bf16* __restrict__ ewb,
                           float* __restrict__ De) {
  __shared__ float in[256];
  __shared__ float h[128];
  __shared__ float e2[128];
  int t = threadIdx.x;
  long e = blockIdx.x;
  in[t]       = edges[e * 128 + t];
  in[128 + t] = upd[e * 128 + t];
  if (e == 0) { De[t] = 0.f; De[128 + t] = 0.f; }
  __syncthreads();
  float a = b1[t];
#pragma unroll 4
  for (int k = 0; k < 256; ++k) a += in[k] * w1[t * 256 + k];
  h[t] = fmaxf(a, 0.f);
  __syncthreads();
  float v = b2[t];
#pragma unroll 4
  for (int k = 0; k < 128; ++k) v += h[k] * w2[t * 128 + k];
  e2[t] = v;
  __syncthreads();
  float kk = bk[t];
#pragma unroll 4
  for (int k = 0; k < 128; ++k) kk += e2[k] * Wk[t * 128 + k];
  k2[e * 128 + t] = (__bf16)fmaxf(kk, 0.f);
  float ew = conv_b[t];
#pragma unroll 4
  for (int k = 0; k < 128; ++k) ew += e2[k] * conv_w[k * 128 + t];   // conv_w (D, OUT)
  ewb[e * 128 + t] = (__bf16)ew;
}

// ---------------------------------------------------------------------------
// K7: dots_v = q2 @ k2^T * SCALE; H = top8-masked softmax renormed;
//     Dv = row sums; De += column sums.  grid(N/16), block(256)
__global__ void k_hyper(const __bf16* __restrict__ q2, const __bf16* __restrict__ k2,
                        float* __restrict__ outH, float* __restrict__ Dv,
                        float* __restrict__ De) {
  __shared__ __align__(32) __bf16 smQ[4 * FRAG];        // 16x128 A
  __shared__ __align__(32) __bf16 smK[64 * FRAG];       // 128x256 B, 16 nt x 4 kc
  __shared__ float smD[16 * 256];
  int t = threadIdx.x, lane = t & 31, w = t >> 5;
  long r0 = (long)blockIdx.x * 16;
  {  // A fill (vectorized 8-elem groups)
    int m = t >> 4, gi = t & 15;
    v8bf val = *(const v8bf*)(q2 + (r0 + m) * 128 + gi * 8);
    int ln = m + ((gi & 2) ? 16 : 0);
    int h  = (gi & 1) * 8;
    *(v8bf*)(smQ + (gi >> 2) * FRAG + ln * 16 + h) = val;
  }
  // B fill: k2 rows K-contiguous -> 32B gl + 32B ds per 16 elems
#pragma unroll
  for (int i = t; i < 2048; i += 256) {
    int e = i >> 3, gi = i & 7;
    v16bf val = *(const v16bf*)(k2 + e * 128 + gi * 16);
    int ln = (e & 15) + ((gi & 1) ? 16 : 0);
    *(v16bf*)(smK + ((e >> 4) * 4 + (gi >> 1)) * FRAG + ln * 16) = val;
  }
  __syncthreads();
  v16bf afr[4];
#pragma unroll
  for (int kc = 0; kc < 4; ++kc) afr[kc] = load_frag(smQ + kc * FRAG, lane);
  for (int ct = w; ct < 16; ct += 8) {
    v8f acc = {};
#pragma unroll
    for (int kc = 0; kc < 4; ++kc)
      acc = wmma_bf16(afr[kc], load_frag(smK + (ct * 4 + kc) * FRAG, lane), acc);
    int n = ct * 16 + (lane & 15), mb = (lane & 16) ? 8 : 0;
#pragma unroll
    for (int rr = 0; rr < 8; ++rr)
      smD[(rr + mb) * 256 + n] = acc[rr] * SCL;
  }
  __syncthreads();

  for (int rr = 0; rr < 2; ++rr) {
    int m = w * 2 + rr;
    float d[8];
#pragma unroll
    for (int j = 0; j < 8; ++j) d[j] = smD[m * 256 + lane + 32 * j];
    float mx = -3.4e38f;
#pragma unroll
    for (int j = 0; j < 8; ++j) mx = fmaxf(mx, d[j]);
    mx = wave_max(mx);
    float se = 0.f;
#pragma unroll
    for (int j = 0; j < 8; ++j) se += __expf(d[j] - mx);
    se = wave_sum(se);
    float inv = 1.f / se;
    float cur = 3.4e38f;
    for (int it = 0; it < 8; ++it) {                    // 8th-largest extraction
      float cm = -3.4e38f;
#pragma unroll
      for (int j = 0; j < 8; ++j) if (d[j] < cur) cm = fmaxf(cm, d[j]);
      cur = wave_max(cm);
    }
    float thr = cur;
    float hv[8]; float ksum = 0.f;
#pragma unroll
    for (int j = 0; j < 8; ++j) {
      float s = __expf(d[j] - mx) * inv;
      hv[j] = (d[j] >= thr) ? s : 0.f;
      ksum += hv[j];
    }
    ksum = wave_sum(ksum);
    float norm = 1.f / (ksum + 1e-9f);
    float rowsum = 0.f;
#pragma unroll
    for (int j = 0; j < 8; ++j) {
      float Hv = hv[j] * norm;
      outH[(r0 + m) * 256 + lane + 32 * j] = Hv;
      smD[m * 256 + lane + 32 * j] = Hv;
      rowsum += Hv;
    }
    rowsum = wave_sum(rowsum);
    if (lane == 0) Dv[r0 + m] = rowsum;
  }
  __syncthreads();
  float cs = 0.f;
#pragma unroll 4
  for (int m = 0; m < 16; ++m) cs += smD[m * 256 + t];
  atomicAdd(&De[t], cs);
}

// ---------------------------------------------------------------------------
// K8: x_out = (H/De/sqrt(Dv)) @ ew + x @ lin_w^T.  grid(N/16), block(256)
__global__ void k_final(const float* __restrict__ H, const __bf16* __restrict__ xb,
                        const __bf16* __restrict__ ewb, const float* __restrict__ lin_w,
                        const float* __restrict__ De, const float* __restrict__ Dv,
                        float* __restrict__ out) {
  __shared__ __align__(32) __bf16 smEW[64 * FRAG];      // 256x128 B, 8 nt x 8 kc
  __shared__ __align__(32) __bf16 smLW[32 * FRAG];      // 128x128 B, 8 nt x 4 kc
  __shared__ __align__(32) __bf16 smH[8 * FRAG];        // 16x256 A, 8 kc
  __shared__ __align__(32) __bf16 smX[4 * FRAG];        // 16x128 A, 4 kc
  __shared__ float invDe[256];
  __shared__ float invDv[16];
  int t = threadIdx.x, lane = t & 31, w = t >> 5;
  long r0 = (long)blockIdx.x * 16;
  invDe[t] = 1.f / (De[t] + 1e-9f);
  if (t < 16) invDv[t] = rsqrtf(Dv[r0 + t] + 1e-9f);
  // smEW: ewb rows are N-contiguous (K-minor) -> vector gl read, scatter ds
#pragma unroll
  for (int i = t; i < 4096; i += 256) {
    int e = i >> 4, cg = i & 15;
    v8bf val = *(const v8bf*)(ewb + e * 128 + cg * 8);
    int nt = cg >> 1;
    int kc = e >> 5;
    int h  = e & 15;
    int lane0 = (cg & 1) * 8 + ((e & 16) ? 16 : 0);
    __bf16* dst = smEW + (nt * 8 + kc) * FRAG + h;
#pragma unroll
    for (int j = 0; j < 8; ++j) dst[(lane0 + j) * 16] = val[j];
  }
  // smLW: lin_w rows K-contiguous -> full vector fill
#pragma unroll
  for (int i = t; i < 1024; i += 256) {
    int n = i >> 3, gi = i & 7;
    int ln = (n & 15) + ((gi & 1) ? 16 : 0);
    b_fill16_f32(smLW + ((n >> 4) * 4 + (gi >> 1)) * FRAG + ln * 16,
                 lin_w + n * 128 + gi * 16);
  }
  {  // smX: A fill (vectorized)
    int m = t >> 4, gi = t & 15;
    v8bf val = *(const v8bf*)(xb + (r0 + m) * 128 + gi * 8);
    int ln = m + ((gi & 2) ? 16 : 0);
    int h  = (gi & 1) * 8;
    *(v8bf*)(smX + (gi >> 2) * FRAG + ln * 16 + h) = val;
  }
  __syncthreads();
  // smH: H rows K-contiguous; scale + cvt + 16B ds store per 8 elems
#pragma unroll
  for (int i = t; i < 512; i += 256) {
    int m = i >> 5, gi = i & 31;
    const v4f* src = (const v4f*)(H + (r0 + m) * 256 + gi * 8);
    v4f lo = src[0], hi = src[1];
    v8bf val;
#pragma unroll
    for (int j = 0; j < 4; ++j) {
      val[j]     = (__bf16)(lo[j] * invDe[gi * 8 + j]     * invDv[m]);
      val[4 + j] = (__bf16)(hi[j] * invDe[gi * 8 + 4 + j] * invDv[m]);
    }
    int ln = m + ((gi & 2) ? 16 : 0);
    int h  = (gi & 1) * 8;
    *(v8bf*)(smH + (gi >> 2) * FRAG + ln * 16 + h) = val;
  }
  __syncthreads();
  v8f acc = {};
#pragma unroll
  for (int kc = 0; kc < 8; ++kc)       // nodes = H_norm @ ew (K=256)
    acc = wmma_bf16(load_frag(smH + kc * FRAG, lane),
                    load_frag(smEW + (w * 8 + kc) * FRAG, lane), acc);
#pragma unroll
  for (int kc = 0; kc < 4; ++kc)       // + x @ lin_w^T (K=128)
    acc = wmma_bf16(load_frag(smX + kc * FRAG, lane),
                    load_frag(smLW + (w * 4 + kc) * FRAG, lane), acc);
  int n = w * 16 + (lane & 15), mb = (lane & 16) ? 8 : 0;
#pragma unroll
  for (int rr = 0; rr < 8; ++rr)
    out[(r0 + rr + mb) * 128 + n] = acc[rr];
}

// ---------------------------------------------------------------------------
extern "C" void kernel_launch(void* const* d_in, const int* in_sizes, int n_in,
                              void* d_out, int out_size, void* d_ws, size_t ws_size,
                              hipStream_t stream) {
  (void)in_sizes; (void)n_in; (void)out_size; (void)ws_size;
  const float* x        = (const float*)d_in[0];
  const float* noise    = (const float*)d_in[1];
  const float* edges_mu = (const float*)d_in[2];
  const float* edges_ls = (const float*)d_in[3];
  const float* Wq = (const float*)d_in[4];
  const float* bq = (const float*)d_in[5];
  const float* Wk = (const float*)d_in[6];
  const float* bk = (const float*)d_in[7];
  const float* Wv = (const float*)d_in[8];
  const float* bv = (const float*)d_in[9];
  const float* w1 = (const float*)d_in[10];
  const float* b1 = (const float*)d_in[11];
  const float* w2 = (const float*)d_in[12];
  const float* b2 = (const float*)d_in[13];
  const float* ln_in_g = (const float*)d_in[14];
  const float* ln_in_b = (const float*)d_in[15];
  const float* ln_e_g  = (const float*)d_in[16];
  const float* ln_e_b  = (const float*)d_in[17];
  const float* conv_w  = (const float*)d_in[18];
  const float* conv_b  = (const float*)d_in[19];
  const float* lin_w   = (const float*)d_in[20];
  // k_n=32, k_e=8 are compile-time constants of the reference setup.

  float* out_x    = (float*)d_out;
  float* out_H    = out_x + NN * 128;
  float* out_dots = out_H + NN * 256;

  char* ws = (char*)d_ws;
  constexpr size_t SZ_ND  = (size_t)NN * DD * 2;     // bf16 N x 128
  constexpr size_t SZ_ED2 = (size_t)EE * DD * 2;     // bf16 E x 128
  constexpr size_t SZ_ED4 = (size_t)EE * DD * 4;     // f32  E x 128
  size_t off = 0;
  __bf16* kf_b = (__bf16*)(ws + off); off += SZ_ND;
  __bf16* vf_b = (__bf16*)(ws + off); off += SZ_ND;
  __bf16* q2_b = (__bf16*)(ws + off); off += SZ_ND;
  __bf16* x_b  = (__bf16*)(ws + off); off += SZ_ND;
  __bf16* q_b  = (__bf16*)(ws + off); off += SZ_ED2;
  __bf16* k2_b = (__bf16*)(ws + off); off += SZ_ED2;
  __bf16* ew_b = (__bf16*)(ws + off); off += SZ_ED2;
  float* edges_f = (float*)(ws + off); off += SZ_ED4;
  float* upd_f   = (float*)(ws + off); off += SZ_ED4;
  float* stats_f = (float*)(ws + off); off += (size_t)EE * 4 * 4;
  float* De_f    = (float*)(ws + off); off += (size_t)EE * 4;
  float* Dv_f    = (float*)(ws + off); off += (size_t)NN * 4;

  k_edge_prep<<<EE, 128, 0, stream>>>(edges_mu, edges_ls, noise, ln_e_g, ln_e_b,
                                      Wq, bq, edges_f, q_b);
  k_node_gemm<<<NN / 16, 256, 0, stream>>>(x, Wk, bk, Wv, bv, Wq, bq,
                                           ln_in_g, ln_in_b, kf_b, vf_b, q2_b, x_b);
  k_dots<<<dim3((NN + 127) / 128, EE / 16), 256, 0, stream>>>(q_b, kf_b, out_dots);
  k_row_stats<<<EE, 256, 0, stream>>>(out_dots, stats_f);
  k_updates<<<EE / 16, 256, 0, stream>>>(out_dots, stats_f, vf_b, upd_f);
  k_edge_mlp<<<EE, 128, 0, stream>>>(edges_f, upd_f, w1, b1, w2, b2, Wk, bk,
                                     conv_w, conv_b, k2_b, ew_b, De_f);
  k_hyper<<<NN / 16, 256, 0, stream>>>(q2_b, k2_b, out_H, Dv_f, De_f);
  k_final<<<NN / 16, 256, 0, stream>>>(out_H, x_b, ew_b, lin_w, De_f, Dv_f, out_x);
}
